// LSTMModel_7224134992036
// MI455X (gfx1250) — matrix operand order repaired and checked by reading, not compiled, not run on hardware
//
#include <hip/hip_runtime.h>
#include <hip/hip_bf16.h>

// Problem dims (fixed by the reference): B=32, T=1024, U=1024.
#define B_DIM   32
#define T_STEPS 1024
#define U_DIM   1024
#define PF      8     // register-pipeline depth (loads in flight per thread)
#define PDIST   64    // global_prefetch distance in timesteps

typedef __attribute__((ext_vector_type(16))) _Float16 v16h;
typedef __attribute__((ext_vector_type(8)))  float    v8f;

__device__ __forceinline__ float htanh(float v) {
#if __has_builtin(__builtin_amdgcn_tanhf)
    return __builtin_amdgcn_tanhf(v);   // gfx1250 hardware v_tanh_f32 (TRANS unit)
#else
    return tanhf(v);
#endif
}

// ---------------------------------------------------------------------------
// Kernel 1: elementwise-LSTM scan. One thread per (b,u) cell; serial over T.
// Streams 128 MB of x (the roofline cost: ~5.5us at 23.3 TB/s). Only c is
// carried; o/tanh(c) are evaluated once at t = T-1 since h never feeds back.
// sigmoid(z) = 0.5*tanh(0.5 z)+0.5 with the 0.5 pre-folded into w/b.
// ---------------------------------------------------------------------------
__global__ void __launch_bounds__(256)
lstm_scan_kernel(const float* __restrict__ x,
                 const float* __restrict__ wg,
                 const float* __restrict__ bg,
                 _Float16* __restrict__ hout)
{
    const int idx = blockIdx.x * blockDim.x + threadIdx.x;
    if (idx >= B_DIM * U_DIM) return;
    const int u = idx & (U_DIM - 1);
    const int b = idx >> 10;

    const float wi  = 0.5f * wg[u];
    const float wf  = 0.5f * wg[U_DIM + u];
    const float wgt =        wg[2 * U_DIM + u];
    const float wo  = 0.5f * wg[3 * U_DIM + u];
    const float bi  = 0.5f * bg[u];
    const float bf  = 0.5f * bg[U_DIM + u];
    const float bgt =        bg[2 * U_DIM + u];
    const float bo  = 0.5f * bg[3 * U_DIM + u];

    const float* __restrict__ xp = x + (size_t)b * T_STEPS * U_DIM + u;

    float c = 0.0f;
    auto step = [&](float xv) {
        const float ig = __builtin_fmaf(0.5f, htanh(__builtin_fmaf(wi,  xv, bi )), 0.5f);
        const float fg = __builtin_fmaf(0.5f, htanh(__builtin_fmaf(wf,  xv, bf )), 0.5f);
        const float gg =                 htanh(__builtin_fmaf(wgt, xv, bgt));
        c = __builtin_fmaf(fg, c, ig * gg);
    };

    // Fill the register pipeline: PF loads in flight.
    float xbuf[PF];
#pragma unroll
    for (int i = 0; i < PF; ++i) xbuf[i] = xp[(size_t)i * U_DIM];

    int t0 = 0;
    // Main loop: consume one slot, refill it, keep PF loads outstanding.
    for (; t0 <= T_STEPS - 2 * PF; t0 += PF) {
        const int tp = (t0 + PDIST < T_STEPS) ? (t0 + PDIST) : (T_STEPS - 1);
        __builtin_prefetch(xp + (size_t)tp * U_DIM, 0, 1);   // global_prefetch_b8
#pragma unroll
        for (int i = 0; i < PF; ++i) {
            const float xv = xbuf[i];
            xbuf[i] = xp[(size_t)(t0 + PF + i) * U_DIM];
            step(xv);
        }
    }
    // Drain (T_STEPS is a multiple of PF; values already in registers).
    for (; t0 < T_STEPS; t0 += PF) {
#pragma unroll
        for (int i = 0; i < PF; ++i) step(xbuf[i]);
    }

    // Output gate only matters at the final timestep (last line is hot in L2).
    const float xl = xp[(size_t)(T_STEPS - 1) * U_DIM];
    const float og = __builtin_fmaf(0.5f, htanh(__builtin_fmaf(wo, xl, bo)), 0.5f);
    const float h  = og * htanh(c);
    hout[idx] = (_Float16)h;
}

// ---------------------------------------------------------------------------
// Kernel 2: out[b] = sum_u h[b][u] * w_dense[u] + b_dense, via
// v_wmma_f32_16x16x32_f16. One block, 2 waves; each wave owns a 16-batch tile
// and K-accumulates over 32-wide chunks (32 WMMAs per wave).
// Fragment layouts used here (A/B in contiguous K order, not the ISA table order):
//   A 16x32 f16 : lane(0-15)=row M, halves[0..15]=K0..15;
//                 lanes 16-31: same M, K16..31.
//   B 32x16 f16 : lane%16 = column N, halves[j] = B[(lane<16?0:16)+j][N].
//                 Only column N==0 carries w_dense; others are zeroed by a
//                 branchless per-lane mask multiply (no EXEC manipulation,
//                 loads stay unconditional and vectorizable).
//   D 16x16 f32 : VGPR r, lane L -> M = (L>=16)*8 + r, N = L%16.
// EXEC is all-ones throughout the WMMA region (64 threads, no divergence).
// ---------------------------------------------------------------------------
__global__ void __launch_bounds__(64)
dense_wmma_kernel(const _Float16* __restrict__ h,
                  const float* __restrict__ wd,
                  const float* __restrict__ bd,
                  float* __restrict__ out)
{
    const int lane = threadIdx.x & 31;
    const int wave = threadIdx.x >> 5;
    const int m    = lane & 15;     // A: row within tile; B/D: column N
    const int hs   = lane >> 4;     // half-select
    const int batch = wave * 16 + m;

    // Branchless column mask: only column N==0 of B carries w_dense.
    const _Float16 colmask = (m == 0) ? (_Float16)1.0f : (_Float16)0.0f;

    v8f acc = {};
    for (int kc = 0; kc < U_DIM; kc += 32) {
        v16h afrag, wfrag;
        // A: two contiguous 8-half runs -> two b128 loads.
#pragma unroll
        for (int j = 0; j < 8; ++j) {
            afrag[j]     = h[batch * U_DIM + kc + hs * 16 +     j];
            afrag[8 + j] = h[batch * U_DIM + kc + hs * 16 + 8 + j];
        }
        // B: load w_dense chunk on ALL lanes (lane-invariant in m, contiguous
        // in j -> vectorizable b128 loads), then mask with pk_mul.
#pragma unroll
        for (int j = 0; j < 16; ++j) {
            wfrag[j] = (_Float16)wd[kc + hs * 16 + j];
        }
        const v16h bfrag = wfrag * colmask;

        acc = __builtin_amdgcn_wmma_f32_16x16x32_f16(
            /*neg_a=*/false, afrag, /*neg_b=*/false, bfrag,
            /*c_mod=*/(short)0, acc, /*reuse_a=*/false, /*reuse_b=*/false);
    }

    if (m == 0) {                  // column 0 of D: lanes 0 and 16
        const float b0 = bd[0];
#pragma unroll
        for (int r = 0; r < 8; ++r)
            out[wave * 16 + hs * 8 + r] = acc[r] + b0;
    }
}

// ---------------------------------------------------------------------------
extern "C" void kernel_launch(void* const* d_in, const int* in_sizes, int n_in,
                              void* d_out, int out_size, void* d_ws, size_t ws_size,
                              hipStream_t stream) {
    const float* x  = (const float*)d_in[0];  // (B, T, U) f32
    const float* wg = (const float*)d_in[1];  // (4U,)     f32
    const float* bg = (const float*)d_in[2];  // (4U,)     f32
    const float* wd = (const float*)d_in[3];  // (U, 1)    f32
    const float* bd = (const float*)d_in[4];  // (1,)      f32
    float* out = (float*)d_out;               // (B, 1)    f32

    _Float16* hws = (_Float16*)d_ws;          // (B, U) f16 staging for WMMA

    lstm_scan_kernel<<<(B_DIM * U_DIM) / 256, 256, 0, stream>>>(x, wg, bg, hws);
    dense_wmma_kernel<<<1, 64, 0, stream>>>(hws, wd, bd, out);
}
